// EDM_42013370090070
// MI455X (gfx1250) — matrix operand order repaired and checked by reading, not compiled, not run on hardware
//
#include <hip/hip_runtime.h>

typedef __attribute__((ext_vector_type(2)))  _Float16 v2h;
typedef __attribute__((ext_vector_type(8)))  _Float16 v8h;
typedef __attribute__((ext_vector_type(16))) _Float16 v16h;
typedef __attribute__((ext_vector_type(8)))  float    v8f;

#define C_DIM  1000        // classes
#define C_PAD  1024        // K padded to multiple of 32 (zero-filled)
#define ROW_H  1032        // LDS row stride in halves (2064 B, 16B-aligned, bank-skewed)

__global__ __launch_bounds__(32)
void EDM_42013370090070_main(const float* __restrict__ o0, const float* __restrict__ o1,
                             const float* __restrict__ o2, const float* __restrict__ o3,
                             float* __restrict__ out, int nRows)
{
    __shared__ __align__(16) _Float16 sE[16 * ROW_H];  // 16 vectors x 1024 f16 (padded)
    __shared__ float sG[16][17];                       // 16x16 Gram (padded)

    const int lane = threadIdx.x;        // wave32: 0..31
    const int row0 = blockIdx.x * 4;     // 4 rows per wave
    if (row0 + 4 > nRows) return;

    // ---------- Phase 1: load once, row-max, e = exp(x - max) -> f16 in LDS ----------
    #pragma unroll
    for (int v = 0; v < 16; ++v) {
        const int mdl = v & 3;
        const float* src = (mdl == 0) ? o0 : (mdl == 1) ? o1 : (mdl == 2) ? o2 : o3;
        src += (size_t)(row0 + (v >> 2)) * C_DIM;

        float xs0[16], xs1[16];
        float m = -1e30f;
        #pragma unroll
        for (int i = 0; i < 16; ++i) {
            const int c2 = i * 64 + 2 * lane;          // even; pairs never straddle 1000
            if (c2 < C_DIM) {
                const float2 t = *(const float2*)(src + c2);   // coalesced b64
                xs0[i] = t.x; xs1[i] = t.y;
                m = fmaxf(m, fmaxf(t.x, t.y));
            } else { xs0[i] = 0.f; xs1[i] = 0.f; }
        }
        #pragma unroll
        for (int s = 16; s >= 1; s >>= 1) m = fmaxf(m, __shfl_xor(m, s));

        _Float16* rowp = &sE[v * ROW_H];
        #pragma unroll
        for (int i = 0; i < 16; ++i) {
            const int c2 = i * 64 + 2 * lane;          // max 1022 -> fills pad with 0
            v2h h = { (_Float16)0.f, (_Float16)0.f };
            if (c2 < C_DIM) {
                h.x = (_Float16)__expf(xs0[i] - m);    // softmax 1/sum cancels in cosine
                h.y = (_Float16)__expf(xs1[i] - m);
            }
            *(v2h*)(rowp + c2) = h;
        }
    }
    __syncthreads();

    // ---------- Phase 2: 16x16 Gram over K=1024 via v_wmma_f32_16x16x32_f16 ----------
    const int g = lane >> 4;             // lane group (K-offset select)
    const int r = lane & 15;             // fragment row / column index
    const _Float16* rp = &sE[r * ROW_H];
    v8f acc = {0.f, 0.f, 0.f, 0.f, 0.f, 0.f, 0.f, 0.f};

    #pragma unroll
    for (int kc = 0; kc < C_PAD / 32; ++kc) {
        const int kb = kc * 32;
        // A (16x32 f16): lane<16 -> K 0..15 ; lane>=16 -> K 16..31 (contiguous, matches B)
        v8h a0 = *(const v8h*)(rp + kb + g * 16);
        v8h a1 = *(const v8h*)(rp + kb + g * 16 + 8);
        v16h A;
        #pragma unroll
        for (int t = 0; t < 8; ++t) { A[t] = a0[t]; A[t + 8] = a1[t]; }
        // B (32x16 f16): lane L holds vector N=L&15, 16 contiguous K at +(L>=16)*16
        v16h B = *(const v16h*)(rp + kb + g * 16);
        acc = __builtin_amdgcn_wmma_f32_16x16x32_f16(false, A, false, B,
                                                     (short)0, acc, false, false);
    }
    #pragma unroll
    for (int t = 0; t < 8; ++t) sG[t + g * 8][r] = acc[t];   // M = t+8g, N = r
    __syncthreads();

    // ---------- Phase 3: cosine sims of 6 pairs per row + logsumexp ----------
    float lse = 0.f;
    if (lane < 4) {
        const int b = lane * 4;          // 4x4 diagonal block of the Gram
        const float i0 = rsqrtf(sG[b+0][b+0]);
        const float i1 = rsqrtf(sG[b+1][b+1]);
        const float i2 = rsqrtf(sG[b+2][b+2]);
        const float i3 = rsqrtf(sG[b+3][b+3]);
        // norms >= 1 (max element -> e=1), so the 1e-6 clamp is provably inactive
        const float d01 = sG[b+0][b+1] * i0 * i1;
        const float d02 = sG[b+0][b+2] * i0 * i2;
        const float d03 = sG[b+0][b+3] * i0 * i3;
        const float d12 = sG[b+1][b+2] * i1 * i2;
        const float d13 = sG[b+1][b+3] * i1 * i3;
        const float d23 = sG[b+2][b+3] * i2 * i3;
        float mx = fmaxf(fmaxf(fmaxf(d01, d02), fmaxf(d03, d12)), fmaxf(d13, d23));
        float s  = __expf(d01 - mx) + __expf(d02 - mx) + __expf(d03 - mx)
                 + __expf(d12 - mx) + __expf(d13 - mx) + __expf(d23 - mx);
        lse = mx + __logf(s);
    }
    #pragma unroll
    for (int s = 16; s >= 1; s >>= 1) lse += __shfl_xor(lse, s);
    if (lane == 0) atomicAdd(out, lse / (float)nRows);
}

__global__ void EDM_42013370090070_zero(float* out)
{
    if (threadIdx.x == 0 && blockIdx.x == 0) out[0] = 0.f;
}

extern "C" void kernel_launch(void* const* d_in, const int* in_sizes, int n_in,
                              void* d_out, int out_size, void* d_ws, size_t ws_size,
                              hipStream_t stream)
{
    const float* o0 = (const float*)d_in[0];
    const float* o1 = (const float*)d_in[1];
    const float* o2 = (const float*)d_in[2];
    const float* o3 = (const float*)d_in[3];
    float* out = (float*)d_out;

    const int nRows = in_sizes[0] / C_DIM;      // 16384
    const int blocks = nRows / 4;               // 4 rows per wave32 workgroup

    EDM_42013370090070_zero<<<1, 32, 0, stream>>>(out);
    EDM_42013370090070_main<<<blocks, 32, 0, stream>>>(o0, o1, o2, o3, out, nRows);
}
